// TopkRoutingMoE_21019569946911
// MI455X (gfx1250) — matrix operand order repaired and checked by reading, not compiled, not run on hardware
//
#include <hip/hip_runtime.h>
#include <cstdint>
#include <cstddef>

// ---------------- problem constants ----------------
#define DIM    1024
#define HID    2816
#define NEXP   8
#define NTOK   8192          // B*T = 4*2048
#define CHUNK  128
#define NCHUNK (HID / CHUNK) // 22
#define MT     32            // token tile per block
#define MAX_TILES 520        // <= 16384/32 + 8

typedef _Float16 f16;
typedef __attribute__((ext_vector_type(16))) _Float16 v16h;
typedef __attribute__((ext_vector_type(8)))  float    v8f;

union FragH { v16h v; uint4 q[2]; };

// ---------------- workspace layout (bytes) ----------------
constexpr size_t SZ_XH  = (size_t)NTOK * DIM * 2;        // 16 MB
constexpr size_t SZ_W   = (size_t)NEXP * HID * DIM * 2;  // 44 MB each
constexpr size_t OFF_XH = 0;
constexpr size_t OFF_W1 = OFF_XH + SZ_XH;
constexpr size_t OFF_W2 = OFF_W1 + SZ_W;
constexpr size_t OFF_W3 = OFF_W2 + SZ_W;
constexpr size_t OFF_TI = OFF_W3 + SZ_W;                 // topIdx: NTOK*2 int
constexpr size_t OFF_TP = OFF_TI + (size_t)NTOK * 2 * 4; // topP:   NTOK*2 f32
constexpr size_t OFF_TL = OFF_TP + (size_t)NTOK * 2 * 4; // tokenList: NEXP*NTOK int
constexpr size_t OFF_SP = OFF_TL + (size_t)NEXP * NTOK * 4;
constexpr size_t OFF_CNT= OFF_SP + (size_t)NEXP * NTOK * 4;
constexpr size_t OFF_NT = OFF_CNT + 256;
constexpr size_t OFF_TE = OFF_NT + 256;
constexpr size_t OFF_TB = OFF_TE + 4096;

// ---------------- fp32 -> f16 conversion ----------------
__global__ void cvt_f32_to_f16(const float* __restrict__ src,
                               f16* __restrict__ dst, int n4) {
  int stride = gridDim.x * blockDim.x;
  for (int j = blockIdx.x * blockDim.x + threadIdx.x; j < n4; j += stride) {
    float4 v = reinterpret_cast<const float4*>(src)[j];
    union { f16 h[4]; uint2 u; } o;
    o.h[0] = (f16)v.x; o.h[1] = (f16)v.y; o.h[2] = (f16)v.z; o.h[3] = (f16)v.w;
    reinterpret_cast<uint2*>(dst)[j] = o.u;
  }
}

// ---------------- gate: scores + top-2 + softmax ----------------
__global__ void gate_topk(const float* __restrict__ x, const float* __restrict__ Wg,
                          int* __restrict__ topIdx, float* __restrict__ topP) {
  __shared__ float sWg[NEXP * DIM];
  for (int i = threadIdx.x; i < NEXP * DIM; i += blockDim.x) sWg[i] = Wg[i];
  __syncthreads();
  int t = blockIdx.x * blockDim.x + threadIdx.x;   // grid sized to NTOK exactly
  const float* xr = x + (size_t)t * DIM;
  float acc[NEXP];
  #pragma unroll
  for (int e = 0; e < NEXP; ++e) acc[e] = 0.f;
  for (int d = 0; d < DIM; ++d) {
    float xv = xr[d];
    #pragma unroll
    for (int e = 0; e < NEXP; ++e) acc[e] += xv * sWg[e * DIM + d];
  }
  float b0 = -1e30f; int i0 = 0;
  #pragma unroll
  for (int e = 0; e < NEXP; ++e) if (acc[e] > b0) { b0 = acc[e]; i0 = e; }
  float b1 = -1e30f; int i1 = 0;
  #pragma unroll
  for (int e = 0; e < NEXP; ++e) if (e != i0 && acc[e] > b1) { b1 = acc[e]; i1 = e; }
  float p0 = 1.f / (1.f + __expf(b1 - b0));  // b1 <= b0, safe
  topIdx[2 * t] = i0; topIdx[2 * t + 1] = i1;
  topP [2 * t] = p0; topP [2 * t + 1] = 1.f - p0;
}

// ---------------- deterministic per-expert list build (1 block, 8 waves) ----------------
__global__ void build_lists(const int* __restrict__ topIdx, const float* __restrict__ topP,
                            int* __restrict__ tokenList, float* __restrict__ slotProb,
                            int* __restrict__ counts, int* __restrict__ nTiles,
                            int* __restrict__ tileE, int* __restrict__ tileB) {
  int lane = threadIdx.x & 31;
  int e    = threadIdx.x >> 5;    // wave id = expert id
  unsigned cnt = 0;
  for (int base = 0; base < NTOK; base += 32) {
    int t  = base + lane;
    int i0 = topIdx[2 * t], i1 = topIdx[2 * t + 1];
    bool sel = (i0 == e) || (i1 == e);
    float p  = (i0 == e) ? topP[2 * t] : topP[2 * t + 1];
    unsigned long long bal = __ballot(sel);
    unsigned pre = (unsigned)__popcll(bal & ((1ull << lane) - 1ull));
    if (sel) {
      tokenList[e * NTOK + cnt + pre] = t;
      slotProb [e * NTOK + cnt + pre] = p;
    }
    cnt += (unsigned)__popcll(bal);
  }
  if (lane == 0) counts[e] = (int)cnt;
  unsigned padded = (cnt + (MT - 1)) & ~(unsigned)(MT - 1);
  for (unsigned i = cnt + lane; i < padded; i += 32) tokenList[e * NTOK + i] = -1;
  __syncthreads();
  if (threadIdx.x == 0) {
    int nt = 0;
    for (int ee = 0; ee < NEXP; ++ee)
      for (int b = 0; b < counts[ee]; b += MT) { tileE[nt] = ee; tileB[nt] = b; ++nt; }
    *nTiles = nt;
  }
}

// ---------------- fused SwiGLU expert FFN, WMMA f16 ----------------
__global__ __launch_bounds__(256, 1)
void moe_ffn(const f16* __restrict__ xh, const f16* __restrict__ W1h,
             const f16* __restrict__ W2h, const f16* __restrict__ W3h,
             const int* __restrict__ tokenList, const float* __restrict__ slotProb,
             const int* __restrict__ nTiles, const int* __restrict__ tileE,
             const int* __restrict__ tileB, float* __restrict__ out) {
  if ((int)blockIdx.x >= *nTiles) return;
  const int e  = tileE[blockIdx.x];
  const int s0 = tileB[blockIdx.x];

  __shared__ __align__(16) f16 hlds[MT * (CHUNK + 8)];  // +8 halves pad -> no bank conflicts
  __shared__ int   stok[MT];
  __shared__ float sprob[MT];

  const int lane  = threadIdx.x & 31;
  const int w     = threadIdx.x >> 5;       // 8 waves
  if (threadIdx.x < MT) {
    int gi  = e * NTOK + s0 + threadIdx.x;
    int raw = tokenList[gi];
    stok[threadIdx.x]  = raw;
    sprob[threadIdx.x] = (raw < 0) ? 0.f : slotProb[gi];
  }
  __syncthreads();

  const int jn    = lane & 15;        // A-row / B-col / C-col index within 16
  const int rowHi = (lane >> 4) & 1;
  const int koffA = rowHi * 16;       // A frag: one 16-half run
  const int koffB = rowHi * 16;       // B frag: one 16-half run

  // per-lane gathered token row offsets (elements) for A
  int tok0 = stok[jn];
  int tok1 = stok[16 + jn];
  const size_t offA0 = (size_t)(tok0 < 0 ? 0 : tok0) * DIM;
  const size_t offA1 = (size_t)(tok1 < 0 ? 0 : tok1) * DIM;

  // persistent y accumulators: wave owns d-cols [w*128, w*128+128) for 2 M-blocks
  v8f accY[16];
  #pragma unroll
  for (int i = 0; i < 16; ++i) accY[i] = (v8f){0.f,0.f,0.f,0.f,0.f,0.f,0.f,0.f};

  const f16* w1base = W1h + (size_t)e * HID * DIM;
  const f16* w2base = W2h + (size_t)e * HID * DIM;
  const f16* w3row  = W3h + ((size_t)e * DIM + (size_t)(w * 128 + jn)) * HID;

  for (int c = 0; c < NCHUNK; ++c) {
    // Opaque copies of the A-row OFFSETS (not pointers!): prevents hoisting of
    // the chunk-invariant A-fragment loads (which caused scratch spills), while
    // keeping the loads in the global address space (global_load_b128, pure
    // LOADcnt) since xh's addrspace inference is preserved.
    size_t o0 = offA0, o1 = offA1;
    asm volatile("" : "+v"(o0), "+v"(o1));
    const f16* ra[2] = { xh + o0, xh + o1 };

    // ---- up projection: this wave computes h columns j = c*128 + w*16 + [0,16) ----
    v8f z = (v8f){0.f,0.f,0.f,0.f,0.f,0.f,0.f,0.f};
    v8f u1[2] = { z, z }, u2[2] = { z, z };
    const size_t j = (size_t)(c * CHUNK + w * 16 + jn);
    const f16* w1row = w1base + j * DIM;
    const f16* w2row = w2base + j * DIM;
    for (int kb = 0; kb < DIM; kb += 32) {
      FragH b1, b2;
      const f16* p1 = w1row + kb + koffB;
      b1.q[0] = *(const uint4*)(p1); b1.q[1] = *(const uint4*)(p1 + 8);
      const f16* p2 = w2row + kb + koffB;
      b2.q[0] = *(const uint4*)(p2); b2.q[1] = *(const uint4*)(p2 + 8);
      #pragma unroll
      for (int mb = 0; mb < 2; ++mb) {
        FragH a;
        const f16* pa = ra[mb] + kb + koffA;
        a.q[0] = *(const uint4*)(pa);
        a.q[1] = *(const uint4*)(pa + 8);
        u1[mb] = __builtin_amdgcn_wmma_f32_16x16x32_f16(false, a.v, false, b1.v,
                                                        (short)0, u1[mb], false, false);
        u2[mb] = __builtin_amdgcn_wmma_f32_16x16x32_f16(false, a.v, false, b2.v,
                                                        (short)0, u2[mb], false, false);
      }
    }

    // SwiGLU epilogue -> f16 chunk in LDS
    #pragma unroll
    for (int mb = 0; mb < 2; ++mb) {
      #pragma unroll
      for (int r = 0; r < 8; ++r) {
        float a = u1[mb][r], b = u2[mb][r];
        float hv = (a / (1.f + __expf(-a))) * b;
        int m = mb * 16 + rowHi * 8 + r;
        hlds[m * (CHUNK + 8) + w * 16 + jn] = (f16)hv;
      }
    }
    __syncthreads();

    // ---- down projection: y += h_chunk(32x128) * W3_chunk^T ----
    #pragma unroll
    for (int kk = 0; kk < 4; ++kk) {
      const int kbase = kk * 32;
      FragH aD[2];
      #pragma unroll
      for (int mb = 0; mb < 2; ++mb) {
        const f16* ph = &hlds[(mb * 16 + jn) * (CHUNK + 8) + kbase + koffA];
        aD[mb].q[0] = *(const uint4*)(ph);
        aD[mb].q[1] = *(const uint4*)(ph + 8);
      }
      const int jcol = c * CHUNK + kbase + koffB;
      #pragma unroll
      for (int db = 0; db < 8; ++db) {
        FragH b3;
        const f16* p3 = w3row + (size_t)db * 16 * HID + jcol;
        b3.q[0] = *(const uint4*)(p3); b3.q[1] = *(const uint4*)(p3 + 8);
        #pragma unroll
        for (int mb = 0; mb < 2; ++mb)
          accY[mb * 8 + db] = __builtin_amdgcn_wmma_f32_16x16x32_f16(
              false, aD[mb].v, false, b3.v, (short)0, accY[mb * 8 + db], false, false);
      }
    }
    __syncthreads();
  }

  // ---- weighted scatter into fp32 output ----
  #pragma unroll
  for (int mb = 0; mb < 2; ++mb) {
    #pragma unroll
    for (int r = 0; r < 8; ++r) {
      int m   = mb * 16 + rowHi * 8 + r;
      int tok = stok[m];
      if (tok < 0) continue;
      float pw = sprob[m];
      size_t base = (size_t)tok * DIM + (size_t)(w * 128 + jn);
      #pragma unroll
      for (int db = 0; db < 8; ++db)
        atomicAdd(out + base + db * 16, pw * accY[mb * 8 + db][r]);
    }
  }
}

// ---------------- host launch ----------------
extern "C" void kernel_launch(void* const* d_in, const int* in_sizes, int n_in,
                              void* d_out, int out_size, void* d_ws, size_t ws_size,
                              hipStream_t stream) {
  const float* x  = (const float*)d_in[0];
  const float* Wg = (const float*)d_in[1];
  const float* W1 = (const float*)d_in[2];
  const float* W2 = (const float*)d_in[3];
  const float* W3 = (const float*)d_in[4];
  float* out = (float*)d_out;
  char*  ws  = (char*)d_ws;

  f16* xh  = (f16*)(ws + OFF_XH);
  f16* W1h = (f16*)(ws + OFF_W1);
  f16* W2h = (f16*)(ws + OFF_W2);
  f16* W3h = (f16*)(ws + OFF_W3);
  int*   topIdx    = (int*)  (ws + OFF_TI);
  float* topP      = (float*)(ws + OFF_TP);
  int*   tokenList = (int*)  (ws + OFF_TL);
  float* slotProb  = (float*)(ws + OFF_SP);
  int*   counts    = (int*)  (ws + OFF_CNT);
  int*   nTiles    = (int*)  (ws + OFF_NT);
  int*   tileE     = (int*)  (ws + OFF_TE);
  int*   tileB     = (int*)  (ws + OFF_TB);

  const int nW = NEXP * HID * DIM;
  cvt_f32_to_f16<<<2048, 256, 0, stream>>>(x,  xh,  NTOK * DIM / 4);
  cvt_f32_to_f16<<<2048, 256, 0, stream>>>(W1, W1h, nW / 4);
  cvt_f32_to_f16<<<2048, 256, 0, stream>>>(W2, W2h, nW / 4);
  cvt_f32_to_f16<<<2048, 256, 0, stream>>>(W3, W3h, nW / 4);

  gate_topk<<<NTOK / 256, 256, 0, stream>>>(x, Wg, topIdx, topP);
  build_lists<<<1, 256, 0, stream>>>(topIdx, topP, tokenList, slotProb,
                                     counts, nTiles, tileE, tileB);

  hipMemsetAsync(d_out, 0, (size_t)NTOK * DIM * sizeof(float), stream);
  moe_ffn<<<MAX_TILES, 256, 0, stream>>>(xh, W1h, W2h, W3h, tokenList, slotProb,
                                         nTiles, tileE, tileB, out);
}
